// HybridRetention_59090160058654
// MI455X (gfx1250) — matrix operand order repaired and checked by reading, not compiled, not run on hardware
//
#include <hip/hip_runtime.h>

// ---------------------------------------------------------------------------
// Types for CDNA5 WMMA (gfx1250, wave32)
// ---------------------------------------------------------------------------
typedef __attribute__((ext_vector_type(16))) __bf16 v16bf;
typedef __attribute__((ext_vector_type(8)))  float  v8f;

#define B_      4
#define T_      4096
#define D_      1024
#define H_      16
#define DH_     64
#define CH_     16          // retention chunk length
#define NCH_    (T_ / CH_)  // 256 chunks
#define NSEG_   16          // scan segments
#define SEGLEN_ 256
#define NCHAN_  (B_ * D_)   // 4096 scan channels

static __device__ __forceinline__ v8f wmma_bf(v16bf a, v16bf b, v8f c) {
  // D = A(16x32 bf16) * B(32x16 bf16) + C(16x16 f32)
  return __builtin_amdgcn_wmma_f32_16x16x32_bf16(false, a, false, b,
                                                 (short)0, c, false, false);
}

// DECAY^n for n = 0..16
__device__ const float DPOW[17] = {
  1.0f, 0.9f, 0.81f, 0.729f, 0.6561f, 0.59049f, 0.531441f, 0.4782969f,
  0.43046721f, 0.387420489f, 0.3486784401f, 0.31381059609f,
  0.282429536481f, 0.2541865828329f, 0.22876792454961f,
  0.205891132094649f, 0.1853020188851841f
};

// ---------------------------------------------------------------------------
// Kernel 1: chunked retention.
// Grid: B*H*(DH/16) = 256 blocks, 32 threads (1 wave) each.
// Each wave owns a 64x16 slice of the per-(b,h) state S and marches over
// 256 chunks of 16 timesteps, doing everything on v_wmma_f32_16x16x32_bf16.
// ---------------------------------------------------------------------------
__global__ __launch_bounds__(32)
void retention_chunk_kernel(const float* __restrict__ q,
                            const float* __restrict__ k,
                            const float* __restrict__ v,
                            float* __restrict__ ret) {
  __shared__ float lq[16][68];   // chunk Q rows (16 x 64), padded
  __shared__ float lk[16][68];   // chunk K rows
  __shared__ float lv[16][20];   // chunk V e-slice (16 x 16), padded
  __shared__ float la[16][20];   // decayed attention tile relayout
  __shared__ float sS[64][20];   // state relayout (C-layout -> B-layout)

  const int bid  = blockIdx.x;
  const int es   = bid & 3;          // 16-wide slice of v-dim
  const int h    = (bid >> 2) & 15;
  const int b    = bid >> 6;
  const int lane = threadIdx.x;
  const int lo   = lane & 15;
  const int hi   = lane >> 4;

  // Per-lane decay constants (loop-invariant).
  float maskf[8];
#pragma unroll
  for (int r = 0; r < 8; ++r) {
    int dt = r + 8 * hi - lo;                 // t - s in the 16x16 tile
    maskf[r] = (dt >= 0) ? DPOW[dt] : 0.0f;   // causal decay mask
  }
  const float qscale = DPOW[lo + 1];          // d^{j+1}, j = local row
  const float d16    = 0.1853020188851841f;   // DECAY^16

  const v8f vzero = {0.f, 0.f, 0.f, 0.f, 0.f, 0.f, 0.f, 0.f};
  v8f sacc[4] = {vzero, vzero, vzero, vzero}; // S slice: 64(kk) x 16(e)

  const size_t qk_base = (size_t)b * T_ * D_ + (size_t)h * DH_;
  const size_t v_base  = qk_base + (size_t)es * 16;

  for (int c = 0; c < NCH_; ++c) {
    const int t0 = c * CH_;

    // ---- stage Q,K (16x64) and V slice (16x16) into LDS, coalesced -------
    const float* qb = q + qk_base + (size_t)t0 * D_;
    const float* kb = k + qk_base + (size_t)t0 * D_;
    const float* vb = v + v_base  + (size_t)t0 * D_;
#pragma unroll
    for (int j = 0; j < 8; ++j) {
      int f = j * 32 + lane, row = f >> 4, c4 = f & 15;
      *(float4*)&lq[row][c4 * 4] = *(const float4*)(qb + (size_t)row * D_ + c4 * 4);
      *(float4*)&lk[row][c4 * 4] = *(const float4*)(kb + (size_t)row * D_ + c4 * 4);
    }
#pragma unroll
    for (int j = 0; j < 2; ++j) {
      int f = j * 32 + lane, row = f >> 2, c4 = f & 3;
      *(float4*)&lv[row][c4 * 4] = *(const float4*)(vb + (size_t)row * D_ + c4 * 4);
    }
    __syncthreads();

    // ---- build WMMA fragments --------------------------------------------
    v16bf qa0, qa1, qs0, qs1, kb0, kb1, vfrag, vsfrag, kta0, kta1, kta2, kta3;
#pragma unroll
    for (int i = 0; i < 16; ++i) {
      const int ak = (i < 8 ? i : i + 8) + 8 * hi;   // A-layout K index
      float q0 = lq[lo][ak], q1 = lq[lo][ak + 32];
      qa0[i] = (__bf16)q0;             qa1[i] = (__bf16)q1;
      qs0[i] = (__bf16)(q0 * qscale);  qs1[i] = (__bf16)(q1 * qscale);

      const int bk = (i < 8 ? i : i + 8) + 8 * hi;   // B-layout K index
      kb0[i] = (__bf16)lk[lo][bk];
      kb1[i] = (__bf16)lk[lo][bk + 32];

      // V as B operand: K-dim = s (only 16 valid of 32)
      float vv = (i < 8) ? lv[i + 8 * hi][lo] : 0.0f;
      vfrag[i]  = (__bf16)vv;
      vsfrag[i] = (__bf16)(vv * DPOW[15 - ((i + 8 * hi) & 15)]);       // d^{15-s} for state upd

      // K^T as A operand: M = kk_local, K-dim = s (valid when ak < 16)
      float kt = (ak < 16) ? lk[ak][0] : 0.0f;       // placeholder init
      kt       = (ak < 16) ? lk[ak][lo]      : 0.0f; kta0[i] = (__bf16)kt;
      kt       = (ak < 16) ? lk[ak][16 + lo] : 0.0f; kta1[i] = (__bf16)kt;
      kt       = (ak < 16) ? lk[ak][32 + lo] : 0.0f; kta2[i] = (__bf16)kt;
      kt       = (ak < 16) ? lk[ak][48 + lo] : 0.0f; kta3[i] = (__bf16)kt;
    }

    // ---- A = QK^T, then causal decay mask --------------------------------
    v8f aacc = wmma_bf(qa0, kb0, vzero);
    aacc = wmma_bf(qa1, kb1, aacc);
#pragma unroll
    for (int r = 0; r < 8; ++r) aacc[r] *= maskf[r];

    // ---- relayout A (C-layout -> A-layout) and S_in (C -> B) via LDS -----
#pragma unroll
    for (int r = 0; r < 8; ++r) {
      const int m = r + 8 * hi;
      la[m][lo]      = aacc[r];
      sS[m][lo]      = sacc[0][r];
      sS[16 + m][lo] = sacc[1][r];
      sS[32 + m][lo] = sacc[2][r];
      sS[48 + m][lo] = sacc[3][r];
    }
    __syncthreads();

    v16bf afrag, sb0, sb1;
#pragma unroll
    for (int i = 0; i < 16; ++i) {
      afrag[i] = (i < 8) ? (__bf16)la[lo][i + 8 * hi] : (__bf16)0.0f;
      sb0[i] = (__bf16)sS[(i < 8 ? i : i + 8) + 8 * hi][lo];
      sb1[i] = (__bf16)sS[32 + (i < 8 ? i : i + 8) + 8 * hi][lo];
    }

    // ---- o = A V + (d^{j+1} q_j) S_in ------------------------------------
    v8f oacc = wmma_bf(afrag, vfrag, vzero);
    oacc = wmma_bf(qs0, sb0, oacc);
    oacc = wmma_bf(qs1, sb1, oacc);

    // ---- S = d^16 S + K^T diag(d^{15-s}) V -------------------------------
    {
      v8f c0 = sacc[0], c1 = sacc[1], c2 = sacc[2], c3 = sacc[3];
#pragma unroll
      for (int r = 0; r < 8; ++r) { c0[r] *= d16; c1[r] *= d16; c2[r] *= d16; c3[r] *= d16; }
      sacc[0] = wmma_bf(kta0, vsfrag, c0);
      sacc[1] = wmma_bf(kta1, vsfrag, c1);
      sacc[2] = wmma_bf(kta2, vsfrag, c2);
      sacc[3] = wmma_bf(kta3, vsfrag, c3);
    }

    // ---- store retained output (fp32) ------------------------------------
    float* ob = ret + v_base + (size_t)t0 * D_;
#pragma unroll
    for (int r = 0; r < 8; ++r)
      ob[(size_t)(r + 8 * hi) * D_ + lo] = oacc[r];
    __syncthreads();
  }
}

// ---------------------------------------------------------------------------
// Kernel 2: fused dual GEMM  u = X W_in + b_in ; g = sigmoid(X W_gate + b_g)
// X = retained (16384 x 1024). Block tile 128(M) x 64(N), 8 waves.
// ---------------------------------------------------------------------------
__global__ __launch_bounds__(256)
void gemm_ug_kernel(const float* __restrict__ X,
                    const float* __restrict__ Wi, const float* __restrict__ bi,
                    const float* __restrict__ Wg, const float* __restrict__ bgate,
                    float* __restrict__ U, float* __restrict__ G) {
  __shared__ __bf16 As[128][40];
  __shared__ __bf16 Bi[32][76];
  __shared__ __bf16 Bg[32][76];

  const int m0 = blockIdx.x * 128;
  const int n0 = blockIdx.y * 64;
  const int tid = threadIdx.x;
  const int w = tid >> 5, lane = tid & 31, lo = lane & 15, hi = lane >> 4;
  const int mw = w & 3, nw = w >> 2;          // 4 x 2 wave grid

  const v8f vzero = {0.f, 0.f, 0.f, 0.f, 0.f, 0.f, 0.f, 0.f};
  v8f accU[2][2] = {{vzero, vzero}, {vzero, vzero}};
  v8f accG[2][2] = {{vzero, vzero}, {vzero, vzero}};

  for (int kc = 0; kc < D_; kc += 32) {
    // stage A (128x32 fp32 -> bf16)
#pragma unroll
    for (int j = 0; j < 4; ++j) {
      int f = j * 256 + tid, row = f >> 3, c4 = f & 7;
      float4 x = *(const float4*)(X + (size_t)(m0 + row) * D_ + kc + c4 * 4);
      __bf16 t[4] = {(__bf16)x.x, (__bf16)x.y, (__bf16)x.z, (__bf16)x.w};
      *(uint2*)&As[row][c4 * 4] = *(uint2*)t;
    }
    // stage W_in, W_gate (32x64 each)
#pragma unroll
    for (int j = 0; j < 2; ++j) {
      int f = j * 256 + tid, row = f >> 4, c4 = f & 15;
      float4 a = *(const float4*)(Wi + (size_t)(kc + row) * D_ + n0 + c4 * 4);
      float4 c = *(const float4*)(Wg + (size_t)(kc + row) * D_ + n0 + c4 * 4);
      __bf16 ta[4] = {(__bf16)a.x, (__bf16)a.y, (__bf16)a.z, (__bf16)a.w};
      __bf16 tc[4] = {(__bf16)c.x, (__bf16)c.y, (__bf16)c.z, (__bf16)c.w};
      *(uint2*)&Bi[row][c4 * 4] = *(uint2*)ta;
      *(uint2*)&Bg[row][c4 * 4] = *(uint2*)tc;
    }
    __syncthreads();

    v16bf a[2], fbi[2], fbg[2];
#pragma unroll
    for (int i = 0; i < 16; ++i) {
      const int ak = (i < 8 ? i : i + 8) + 8 * hi;
      a[0][i] = As[mw * 32 + lo][ak];
      a[1][i] = As[mw * 32 + 16 + lo][ak];
      const int bk = (i < 8 ? i : i + 8) + 8 * hi;
      fbi[0][i] = Bi[bk][nw * 32 + lo];
      fbi[1][i] = Bi[bk][nw * 32 + 16 + lo];
      fbg[0][i] = Bg[bk][nw * 32 + lo];
      fbg[1][i] = Bg[bk][nw * 32 + 16 + lo];
    }
#pragma unroll
    for (int s1 = 0; s1 < 2; ++s1)
#pragma unroll
      for (int s2 = 0; s2 < 2; ++s2) {
        accU[s1][s2] = wmma_bf(a[s1], fbi[s2], accU[s1][s2]);
        accG[s1][s2] = wmma_bf(a[s1], fbg[s2], accG[s1][s2]);
      }
    __syncthreads();
  }

#pragma unroll
  for (int s1 = 0; s1 < 2; ++s1)
#pragma unroll
    for (int s2 = 0; s2 < 2; ++s2) {
      const int n = n0 + nw * 32 + s2 * 16 + lo;
      const float bu = bi[n], bgv = bgate[n];
#pragma unroll
      for (int r = 0; r < 8; ++r) {
        const int row = m0 + mw * 32 + s1 * 16 + r + 8 * hi;
        float uu = accU[s1][s2][r] + bu;
        float gg = accG[s1][s2][r] + bgv;
        gg = 1.0f / (1.0f + __expf(-gg));
        U[(size_t)row * D_ + n] = uu;
        G[(size_t)row * D_ + n] = gg;
      }
    }
}

// ---------------------------------------------------------------------------
// Kernels 3-5: segmented scan  h_t = g h_{t-1} + (1-g) u_t
// ---------------------------------------------------------------------------
__global__ __launch_bounds__(256)
void scan_phase1(const float* __restrict__ U, const float* __restrict__ G,
                 float* __restrict__ segP, float* __restrict__ segC) {
  const int gid  = blockIdx.x * 256 + threadIdx.x;      // < 65536
  const int chan = gid & (NCHAN_ - 1);
  const int seg  = gid >> 12;
  const int b = chan >> 10, d = chan & (D_ - 1);
  size_t base = ((size_t)b * T_ + seg * SEGLEN_) * D_ + d;
  float P = 1.0f, C = 0.0f;
#pragma unroll 4
  for (int t = 0; t < SEGLEN_; ++t) {
    float gg = G[base], uu = U[base];
    P *= gg;
    C = gg * C + (1.0f - gg) * uu;
    base += D_;
  }
  segP[(size_t)chan * NSEG_ + seg] = P;
  segC[(size_t)chan * NSEG_ + seg] = C;
}

__global__ __launch_bounds__(256)
void scan_phase2(const float* __restrict__ segP, const float* __restrict__ segC,
                 float* __restrict__ segH) {
  const int chan = blockIdx.x * 256 + threadIdx.x;      // < 4096
  float hin = 0.0f;
#pragma unroll
  for (int s = 0; s < NSEG_; ++s) {
    segH[(size_t)chan * NSEG_ + s] = hin;
    hin = segP[(size_t)chan * NSEG_ + s] * hin + segC[(size_t)chan * NSEG_ + s];
  }
}

__global__ __launch_bounds__(256)
void scan_phase3(float* __restrict__ U, const float* __restrict__ G,
                 const float* __restrict__ segH) {
  const int gid  = blockIdx.x * 256 + threadIdx.x;
  const int chan = gid & (NCHAN_ - 1);
  const int seg  = gid >> 12;
  const int b = chan >> 10, d = chan & (D_ - 1);
  size_t base = ((size_t)b * T_ + seg * SEGLEN_) * D_ + d;
  float hcur = segH[(size_t)chan * NSEG_ + seg];
#pragma unroll 4
  for (int t = 0; t < SEGLEN_; ++t) {
    float gg = G[base], uu = U[base];
    hcur = gg * hcur + (1.0f - gg) * uu;
    U[base] = hcur;                     // h overwrites u in place
    base += D_;
  }
}

// ---------------------------------------------------------------------------
// Kernel 6: out = h W_out + b_out + retained   (residual fused)
// ---------------------------------------------------------------------------
__global__ __launch_bounds__(256)
void gemm_out_kernel(const float* __restrict__ Hm, const float* __restrict__ Wo,
                     const float* __restrict__ bo, const float* __restrict__ Res,
                     float* __restrict__ Out) {
  __shared__ __bf16 As[128][40];
  __shared__ __bf16 Bs[32][76];

  const int m0 = blockIdx.x * 128;
  const int n0 = blockIdx.y * 64;
  const int tid = threadIdx.x;
  const int w = tid >> 5, lane = tid & 31, lo = lane & 15, hi = lane >> 4;
  const int mw = w & 3, nw = w >> 2;

  const v8f vzero = {0.f, 0.f, 0.f, 0.f, 0.f, 0.f, 0.f, 0.f};
  v8f acc[2][2] = {{vzero, vzero}, {vzero, vzero}};

  for (int kc = 0; kc < D_; kc += 32) {
#pragma unroll
    for (int j = 0; j < 4; ++j) {
      int f = j * 256 + tid, row = f >> 3, c4 = f & 7;
      float4 x = *(const float4*)(Hm + (size_t)(m0 + row) * D_ + kc + c4 * 4);
      __bf16 t[4] = {(__bf16)x.x, (__bf16)x.y, (__bf16)x.z, (__bf16)x.w};
      *(uint2*)&As[row][c4 * 4] = *(uint2*)t;
    }
#pragma unroll
    for (int j = 0; j < 2; ++j) {
      int f = j * 256 + tid, row = f >> 4, c4 = f & 15;
      float4 a = *(const float4*)(Wo + (size_t)(kc + row) * D_ + n0 + c4 * 4);
      __bf16 t[4] = {(__bf16)a.x, (__bf16)a.y, (__bf16)a.z, (__bf16)a.w};
      *(uint2*)&Bs[row][c4 * 4] = *(uint2*)t;
    }
    __syncthreads();

    v16bf a[2], fb[2];
#pragma unroll
    for (int i = 0; i < 16; ++i) {
      const int ak = (i < 8 ? i : i + 8) + 8 * hi;
      a[0][i] = As[mw * 32 + lo][ak];
      a[1][i] = As[mw * 32 + 16 + lo][ak];
      const int bk = (i < 8 ? i : i + 8) + 8 * hi;
      fb[0][i] = Bs[bk][nw * 32 + lo];
      fb[1][i] = Bs[bk][nw * 32 + 16 + lo];
    }
#pragma unroll
    for (int s1 = 0; s1 < 2; ++s1)
#pragma unroll
      for (int s2 = 0; s2 < 2; ++s2)
        acc[s1][s2] = wmma_bf(a[s1], fb[s2], acc[s1][s2]);
    __syncthreads();
  }

#pragma unroll
  for (int s1 = 0; s1 < 2; ++s1)
#pragma unroll
    for (int s2 = 0; s2 < 2; ++s2) {
      const int n = n0 + nw * 32 + s2 * 16 + lo;
      const float bb = bo[n];
#pragma unroll
      for (int r = 0; r < 8; ++r) {
        const int row = m0 + mw * 32 + s1 * 16 + r + 8 * hi;
        Out[(size_t)row * D_ + n] =
            acc[s1][s2][r] + bb + Res[(size_t)row * D_ + n];
      }
    }
}

// ---------------------------------------------------------------------------
extern "C" void kernel_launch(void* const* d_in, const int* in_sizes, int n_in,
                              void* d_out, int out_size, void* d_ws, size_t ws_size,
                              hipStream_t stream) {
  (void)in_sizes; (void)n_in; (void)out_size; (void)ws_size;
  const float* q     = (const float*)d_in[0];
  const float* k     = (const float*)d_in[1];
  const float* v     = (const float*)d_in[2];
  const float* Wi    = (const float*)d_in[3];
  const float* bi    = (const float*)d_in[4];
  const float* Wg    = (const float*)d_in[5];
  const float* bgate = (const float*)d_in[6];
  const float* Wo    = (const float*)d_in[7];
  const float* bo    = (const float*)d_in[8];
  float* out = (float*)d_out;

  const size_t NTOK = (size_t)B_ * T_ * D_;   // 16,777,216
  float* ws   = (float*)d_ws;
  float* ret  = ws;                 // retention output (residual source)
  float* U    = ret + NTOK;         // u, later overwritten by h
  float* G    = U + NTOK;           // gate
  float* segP = G + NTOK;
  float* segC = segP + (size_t)NCHAN_ * NSEG_;
  float* segH = segC + (size_t)NCHAN_ * NSEG_;

  // 1) chunked retention -> ret
  retention_chunk_kernel<<<B_ * H_ * (DH_ / 16), 32, 0, stream>>>(q, k, v, ret);

  // 2) u = ret@Win + bi ; g = sigmoid(ret@Wg + bg)
  gemm_ug_kernel<<<dim3((B_ * T_) / 128, D_ / 64), 256, 0, stream>>>(
      ret, Wi, bi, Wg, bgate, U, G);

  // 3-5) segmented scan, h written in place over U
  scan_phase1<<<(NCHAN_ * NSEG_) / 256, 256, 0, stream>>>(U, G, segP, segC);
  scan_phase2<<<NCHAN_ / 256, 256, 0, stream>>>(segP, segC, segH);
  scan_phase3<<<(NCHAN_ * NSEG_) / 256, 256, 0, stream>>>(U, G, segH);

  // 6) out = h@Wo + bo + ret
  gemm_out_kernel<<<dim3((B_ * T_) / 128, D_ / 64), 256, 0, stream>>>(
      U, Wo, bo, ret, out);
}
